// Predictor_12833362280639
// MI455X (gfx1250) — matrix operand order repaired and checked by reading, not compiled, not run on hardware
//
#include <hip/hip_runtime.h>
#include <math.h>

// Problem constants (from reference)
#define NPTS 1024
#define DEMB 128
#define DDIS 64
#define DOUT 127
#define KIN  320   // 2*D + DDIS

typedef __attribute__((ext_vector_type(16))) __bf16          v16bf;
typedef __attribute__((ext_vector_type(8)))  float           v8f;
typedef __attribute__((ext_vector_type(4)))  float           v4f;

// 16-byte store payload with only 4-byte alignment guarantee (rows are 127 floats).
struct __attribute__((packed, aligned(4))) f4u { float x, y, z, w; };

// Native float -> bf16 conversion; backend selects v_cvt_pk_bf16_f32.
__device__ __forceinline__ void cvt4(v16bf& dst, int base, v4f q) {
    dst[base + 0] = (__bf16)q.x;
    dst[base + 1] = (__bf16)q.y;
    dst[base + 2] = (__bf16)q.z;
    dst[base + 3] = (__bf16)q.w;
}

// SiLU with hardware approx reciprocal (v_rcp_f32) instead of the IEEE
// div_scale/div_fmas sequence that __frcp_rn expands to.
__device__ __forceinline__ float silu(float x) {
    return x * __builtin_amdgcn_rcpf(1.0f + __expf(-x));
}

// ---------------------------------------------------------------------------
// Kernel 1: per-row / per-col bias tables (padded to 128 cols).
//   OB[o*128 + c] = sum_k o_emb[o,k] * W[c,k]     + b[c]   (c < 127)
//   DB[d*128 + c] = sum_k d_emb[d,k] * W[c,128+k]          (c < 127)
// ---------------------------------------------------------------------------
__global__ void bias_gemm_kernel(const float* __restrict__ o_emb,
                                 const float* __restrict__ d_emb,
                                 const float* __restrict__ W,
                                 const float* __restrict__ b,
                                 float* __restrict__ OB,
                                 float* __restrict__ DB) {
    const int row = blockIdx.x;          // [0,1024) -> OB, [1024,2048) -> DB
    const int c   = threadIdx.x;         // 0..127
    const bool isO = row < NPTS;
    const int r    = isO ? row : (row - NPTS);
    const v4f* emb = (const v4f*)((isO ? o_emb : d_emb) + (size_t)r * DEMB);
    const v4f* w   = (const v4f*)(W + (size_t)c * KIN + (isO ? 0 : DEMB));
    float acc = 0.0f;
    if (c < DOUT) {
        acc = isO ? b[c] : 0.0f;
        #pragma unroll 8
        for (int k = 0; k < DEMB / 4; ++k) {
            v4f e = emb[k], v = w[k];
            acc = fmaf(e.x, v.x, acc);
            acc = fmaf(e.y, v.y, acc);
            acc = fmaf(e.z, v.z, acc);
            acc = fmaf(e.w, v.w, acc);
        }
    }
    float* dst = isO ? OB : DB;
    dst[(size_t)r * 128 + c] = acc;
}

// ---------------------------------------------------------------------------
// Kernel 2: fused einsum + bias + SiLU with swapped WMMA operand roles:
//   D[c][d] = sum_k Wdis[c,k] * dis[d,k] + (OB[o,c] + DB[d,c])
// A = Wdis tile (M = c, register-resident), B = dis^T tile (N = d, streamed).
// The bias is folded into the WMMA C operand (its layout matches exactly:
// VGPR r -> c, lane -> d), so bias loads feed the WMMA inputs (issued early)
// and the epilogue is pure SiLU + store.
// ---------------------------------------------------------------------------
__global__ void __launch_bounds__(256, 1)
dis_wmma_kernel(const float* __restrict__ dis,
                const float* __restrict__ W,
                const float* __restrict__ OB,
                const float* __restrict__ DB,
                float* __restrict__ out,
                int numTiles, int totalWaves) {
    const int lane = threadIdx.x & 31;
    const int hi   = lane >> 4;     // 0: lanes 0-15, 1: lanes 16-31
    const int ln   = lane & 15;
    const int wave = (blockIdx.x * blockDim.x + threadIdx.x) >> 5;

    // ---- A = Wdis fragments, loaded once (8 c-tiles x 2 K-steps) ----------
    // 16-bit A 16x32 layout: lanes 0-15 (M=lane&15): elems 0-7 = K 0..7,
    // elems 8-15 = K 16..23; lanes 16-31: K 8..15 and K 24..31.
    v16bf Am[8][2];
    #pragma unroll
    for (int ct = 0; ct < 8; ++ct) {
        int c = ct * 16 + ln;
        if (c > DOUT - 1) c = DOUT - 1;     // row 127 absent; results discarded
        const float* wr = W + (size_t)c * KIN + 2 * DEMB;
        #pragma unroll
        for (int ks = 0; ks < 2; ++ks) {
            const v4f* w4a = (const v4f*)(wr + ks * 32 + hi * 8);        // K lo 8
            const v4f* w4b = (const v4f*)(wr + ks * 32 + 16 + hi * 8);   // K hi 8
            v16bf au;
            cvt4(au,  0, w4a[0]);
            cvt4(au,  4, w4a[1]);
            cvt4(au,  8, w4b[0]);
            cvt4(au, 12, w4b[1]);
            Am[ct][ks] = au;
        }
    }

    // ------------------------- grid-stride over d tiles ---------------------
    for (int tile = wave; tile < numTiles; tile += totalWaves) {
        const int rowBase = tile << 4;       // 16 global rows; never crosses o
        const int o       = rowBase >> 10;
        const int dBase   = rowBase & (NPTS - 1);

        // Prefetch next iteration's dis tile (4 KB; 128 B per lane).
        {
            const int nt = tile + totalWaves;
            if (nt < numTiles)
                __builtin_prefetch(dis + ((size_t)nt << 4) * DDIS + lane * 32, 0, 1);
        }

        // ---- B = dis^T fragments: col n = ln -> row d = dBase + ln --------
        // 16-bit B 32x16 layout (matches A): lanes 0-15: elems 0-7 = K 0..7,
        // elems 8-15 = K 16..23; lanes 16-31: K 8..15 and K 24..31.
        const float* brow = dis + (size_t)(rowBase + ln) * DDIS + hi * 8;
        v16bf b0, b1;
        {
            const v4f* p = (const v4f*)brow;        // 16B aligned
            cvt4(b0,  0, __builtin_nontemporal_load(p + 0));   // K = hi*8 + 0..3
            cvt4(b0,  4, __builtin_nontemporal_load(p + 1));
            cvt4(b0,  8, __builtin_nontemporal_load(p + 4));
            cvt4(b0, 12, __builtin_nontemporal_load(p + 5));
            cvt4(b1,  0, __builtin_nontemporal_load(p + 8));   // +32 K
            cvt4(b1,  4, __builtin_nontemporal_load(p + 9));
            cvt4(b1,  8, __builtin_nontemporal_load(p + 12));
            cvt4(b1, 12, __builtin_nontemporal_load(p + 13));
        }

        const int   d    = dBase + ln;                 // fixed per lane
        const int   m    = rowBase + ln;               // global output row
        const v4f*  obp  = (const v4f*)(OB + (size_t)o * 128);
        const v4f*  dbp  = (const v4f*)(DB + (size_t)d * 128);
        float*      orow = out + (size_t)m * DOUT;

        #pragma unroll
        for (int ct = 0; ct < 8; ++ct) {
            // Bias as the WMMA C operand: C[r] = OB[o, cb+r] + DB[d, cb+r].
            const int cb = ct * 16 + hi * 8;
            const int q  = cb / 4;
            v4f c0 = obp[q]     + dbp[q];
            v4f c1 = obp[q + 1] + dbp[q + 1];
            v8f acc = __builtin_shufflevector(c0, c1, 0, 1, 2, 3, 4, 5, 6, 7);

            acc = __builtin_amdgcn_wmma_f32_16x16x32_bf16(
                      false, Am[ct][0], false, b0, (short)0, acc, false, false);
            acc = __builtin_amdgcn_wmma_f32_16x16x32_bf16(
                      false, Am[ct][1], false, b1, (short)0, acc, false, false);

            // C/D: VGPR r -> c = cb + r; lane -> d (fixed). Pure SiLU + store.
            float y0 = silu(acc[0]);
            float y1 = silu(acc[1]);
            float y2 = silu(acc[2]);
            float y3 = silu(acc[3]);
            float y4 = silu(acc[4]);
            float y5 = silu(acc[5]);
            float y6 = silu(acc[6]);
            float y7 = silu(acc[7]);

            if (cb + 7 < DOUT) {                       // full 8-wide store
                *(f4u*)(orow + cb)     = f4u{y0, y1, y2, y3};
                *(f4u*)(orow + cb + 4) = f4u{y4, y5, y6, y7};
            } else {                                   // ct=7, hi=1: c=120..126
                *(f4u*)(orow + cb) = f4u{y0, y1, y2, y3};
                orow[cb + 4] = y4;
                orow[cb + 5] = y5;
                orow[cb + 6] = y6;
            }
        }
    }
}

extern "C" void kernel_launch(void* const* d_in, const int* in_sizes, int n_in,
                              void* d_out, int out_size, void* d_ws, size_t ws_size,
                              hipStream_t stream) {
    const float* o_emb = (const float*)d_in[0];
    const float* d_emb = (const float*)d_in[1];
    const float* dis   = (const float*)d_in[2];
    const float* W     = (const float*)d_in[3];
    const float* b     = (const float*)d_in[4];
    float* out = (float*)d_out;

    float* OB = (float*)d_ws;                 // 1024*128 floats = 512 KB
    float* DB = OB + (size_t)NPTS * 128;      // 1024*128 floats = 512 KB

    // Stage 1: bias tables (OB includes b; DB does not).
    bias_gemm_kernel<<<2 * NPTS, 128, 0, stream>>>(o_emb, d_emb, W, b, OB, DB);

    // Stage 2: fused einsum + bias(C operand) + SiLU.
    const int numTiles   = (NPTS * NPTS) / 16;     // 65536 tiles of 16 rows
    const int blocks     = 1024;                   // 8 waves per block
    const int totalWaves = blocks * (256 / 32);    // 8192 waves, 8 tiles each
    dis_wmma_kernel<<<blocks, 256, 0, stream>>>(dis, W, OB, DB, out,
                                                numTiles, totalWaves);
}
